// MambaBlock_75222057222774
// MI455X (gfx1250) — matrix operand order repaired and checked by reading, not compiled, not run on hardware
//
#include <hip/hip_runtime.h>
#include <hip/hip_bf16.h>

// ---------------------------------------------------------------------------
// Mamba block for gfx1250 (MI455X).
// GEMMs: v_wmma_f32_16x16x32_f16, 32x64 tile per wave (MT=2 x NT=4),
//        compile-time N/K so fragment offsets fold into VMEM immediates.
// Scan: TDM (tensor_load_to_lds) double-buffered B/C broadcast + s_wait_tensorcnt.
// ---------------------------------------------------------------------------

typedef __attribute__((ext_vector_type(16))) _Float16 v16h;
typedef __attribute__((ext_vector_type(8)))  _Float16 v8h;
typedef __attribute__((ext_vector_type(8)))  float    v8f;
typedef __attribute__((ext_vector_type(4)))  unsigned int u32x4;
typedef __attribute__((ext_vector_type(8)))  int      i32x8;
typedef __attribute__((ext_vector_type(4)))  int      i32x4;

#define DMODEL 1024
#define DSTATE 16
#define DCONV  4
#define DINNER 2048
#define DTRANK 64
#define NB     2
#define NL     2048
#define NTOK   (NB * NL)           // 4096 tokens
#define DX     (DTRANK + 2*DSTATE) // 96
#define CHUNK  8                   // scan steps staged per TDM transfer

// ------------------------------- helpers -----------------------------------

__global__ void cast_f32_to_f16_kernel(const float* __restrict__ src,
                                       _Float16* __restrict__ dst, int n) {
    int i = blockIdx.x * blockDim.x + threadIdx.x;
    if (i < n) dst[i] = (_Float16)src[i];
}

// LayerNorm over last dim (1024), one block per token, output f16.
__global__ void layernorm_kernel(const float* __restrict__ x,
                                 const float* __restrict__ gamma,
                                 const float* __restrict__ beta,
                                 _Float16* __restrict__ xn) {
    const int t = blockIdx.x;
    const float* xr = x + (size_t)t * DMODEL;
    __shared__ float red[256];

    float s = 0.f;
    for (int i = threadIdx.x; i < DMODEL; i += 256) s += xr[i];
    red[threadIdx.x] = s;
    __syncthreads();
    for (int off = 128; off > 0; off >>= 1) {
        if (threadIdx.x < off) red[threadIdx.x] += red[threadIdx.x + off];
        __syncthreads();
    }
    const float mu = red[0] * (1.f / DMODEL);
    __syncthreads();

    float v = 0.f;
    for (int i = threadIdx.x; i < DMODEL; i += 256) {
        float d = xr[i] - mu;
        v += d * d;
    }
    red[threadIdx.x] = v;
    __syncthreads();
    for (int off = 128; off > 0; off >>= 1) {
        if (threadIdx.x < off) red[threadIdx.x] += red[threadIdx.x + off];
        __syncthreads();
    }
    const float rs = rsqrtf(red[0] * (1.f / DMODEL) + 1e-5f);
    __syncthreads();

    _Float16* o = xn + (size_t)t * DMODEL;
    for (int i = threadIdx.x; i < DMODEL; i += 256)
        o[i] = (_Float16)((xr[i] - mu) * rs * gamma[i] + beta[i]);
}

// ------------------------------ WMMA GEMM ----------------------------------
// C[M,N] (f32) = A[M,K] (f16, row-major) @ W[N,K]^T (f16, row-major)
//                (+ addend[M,N] if non-null)
// Each wave computes a (16*MT) x (16*NT) tile; B fragments reused MT times.
// N, K compile-time: per-fragment offsets (16*i*K elems) become immediate
// offsets on a single base address per matrix -> no pointer-array spills.
// Requires: M%(16*MT)==0, N%(16*NT)==0, K%32==0, rows 32B aligned.

template <int MT, int NT, int N, int K>
__global__ void gemm_wmma_kernel(const _Float16* __restrict__ A,
                                 const _Float16* __restrict__ W,
                                 float* __restrict__ C,
                                 int M,
                                 const float* __restrict__ addend) {
    const int lane = threadIdx.x & 31;
    const int wave = threadIdx.x >> 5;
    constexpr int ngrp = N / (16 * NT);
    const int grp = blockIdx.x * (blockDim.x >> 5) + wave;
    if (grp >= (M / (16 * MT)) * ngrp) return;  // wave-uniform: EXEC all-1s

    const int mi = grp / ngrp;
    const int nj = grp % ngrp;
    const int m0 = mi * 16 * MT;
    const int n0 = nj * 16 * NT;

    const int m = lane & 15;
    const int g = lane >> 4;

    // A-fragment (contiguous K, matching B):
    //   lanes 0-15: K = 0..15; lanes 16-31: K = 16..31
    const _Float16* abase = A + (size_t)(m0 + m) * K + 16 * g;
    // B-fragment: column n = lane&15 of B == row (n0+n) of W;
    //   lanes 0-15 hold K=0..15, lanes 16-31 hold K=16..31 (contiguous v16h)
    const _Float16* wbase = W + (size_t)(n0 + m) * K + 16 * g;

    v8f acc[MT][NT];
#pragma unroll
    for (int i = 0; i < MT; ++i)
#pragma unroll
        for (int j = 0; j < NT; ++j)
#pragma unroll
            for (int e = 0; e < 8; ++e) acc[i][j][e] = 0.f;

    for (int k0 = 0; k0 < K; k0 += 32) {
        // stage all fragments first so loads can overlap before the WMMA group
        v16h a[MT];
#pragma unroll
        for (int i = 0; i < MT; ++i) {
            v8h lo = *(const v8h*)(abase + i * 16 * K + k0);
            v8h hi = *(const v8h*)(abase + i * 16 * K + k0 + 8);
#pragma unroll
            for (int e = 0; e < 8; ++e) { a[i][e] = lo[e]; a[i][e + 8] = hi[e]; }
        }
        v16h bfr[NT];
#pragma unroll
        for (int j = 0; j < NT; ++j)
            bfr[j] = *(const v16h*)(wbase + j * 16 * K + k0);

        if (k0 + 64 < K) {                       // global_prefetch_b8
            __builtin_prefetch(abase + k0 + 64, 0, 1);
            __builtin_prefetch(wbase + k0 + 64, 0, 1);
        }

#pragma unroll
        for (int i = 0; i < MT; ++i)
#pragma unroll
            for (int j = 0; j < NT; ++j)
                acc[i][j] = __builtin_amdgcn_wmma_f32_16x16x32_f16(
                    false, a[i], false, bfr[j], (short)0, acc[i][j], false, false);
    }

    // D layout: n = lane&15; VGPR r holds M = r + 8*g
    const int n = lane & 15;
#pragma unroll
    for (int i = 0; i < MT; ++i)
#pragma unroll
        for (int j = 0; j < NT; ++j)
#pragma unroll
            for (int r = 0; r < 8; ++r) {
                size_t idx = (size_t)(m0 + 16 * i + r + 8 * g) * N + n0 + 16 * j + n;
                float v = acc[i][j][r];
                if (addend) v += addend[idx];
                C[idx] = v;
            }
}

// --------------------------- conv + SiLU -----------------------------------
__global__ void conv_silu_kernel(const float* __restrict__ u,
                                 const float* __restrict__ cw,
                                 const float* __restrict__ cb,
                                 float* __restrict__ ucf,
                                 _Float16* __restrict__ uch) {
    size_t idx = (size_t)blockIdx.x * blockDim.x + threadIdx.x;
    if (idx >= (size_t)NTOK * DINNER) return;
    const int d = (int)(idx & (DINNER - 1));
    const size_t t = idx >> 11;              // token = b*NL + l
    const int l = (int)(t & (NL - 1));

    float acc = cb[d];
#pragma unroll
    for (int j = 0; j < DCONV; ++j) {
        int ls = l - (DCONV - 1) + j;
        if (ls >= 0)
            acc += cw[d * DCONV + j] * u[(t - (DCONV - 1) + j) * DINNER + d];
    }
    float s = acc / (1.f + __expf(-acc));    // SiLU
    ucf[idx] = s;
    uch[idx] = (_Float16)s;
}

// dt_r slice of x_dbl -> f16 GEMM input
__global__ void dtr_cast_kernel(const float* __restrict__ xdbl,
                                _Float16* __restrict__ dtr) {
    int i = blockIdx.x * blockDim.x + threadIdx.x;
    if (i >= NTOK * DTRANK) return;
    int t = i >> 6, r = i & 63;
    dtr[i] = (_Float16)xdbl[(size_t)t * DX + r];
}

// dt = softplus(dt_lin + b_dt[d]) in place
__global__ void softplus_kernel(float* __restrict__ dt,
                                const float* __restrict__ b_dt) {
    size_t idx = (size_t)blockIdx.x * blockDim.x + threadIdx.x;
    if (idx >= (size_t)NTOK * DINNER) return;
    int d = (int)(idx & (DINNER - 1));
    float x = dt[idx] + b_dt[d];
    dt[idx] = (x > 20.f) ? x : log1pf(__expf(x));
}

// --------------------------- TDM helper ------------------------------------
// Stage `rows` consecutive B/C slices (32 f32 each, row stride 96 f32) from
// x_dbl into LDS via the Tensor Data Mover. D# per CDNA5 ISA ch. 8:
//   group0: count=1 | lds_addr | global_addr | type=2
//   group1: data_size=4B, tensor_dim0=32, tensor_dim0_stride=96,
//           tile_dim0=32, tile_dim1=rows
__device__ inline void tdm_load_bc(const float* gptr, unsigned lds_off, int rows) {
    unsigned long long ga = (unsigned long long)(size_t)gptr;
    u32x4 g0 = {0u, 0u, 0u, 0u};
    g0[0] = 1u;                                     // count=1, user mode
    g0[1] = lds_off;                                // lds_addr (bytes)
    g0[2] = (unsigned)(ga & 0xffffffffu);           // global_addr[31:0]
    g0[3] = (unsigned)((ga >> 32) & 0x1ffffffu)     // global_addr[56:32]
          | (2u << 30);                             // type=2 ("image")
    i32x8 g1 = {0, 0, 0, 0, 0, 0, 0, 0};
    const unsigned td0 = 32u, tile0 = 32u;
    const unsigned rr = (unsigned)rows;
    g1[0] = (int)(2u << 16);                        // data_size = 4 bytes
    g1[1] = (int)((td0 & 0xffffu) << 16);           // tensor_dim0[15:0]
    g1[2] = (int)(((td0 >> 16) & 0xffffu) | ((rr & 0xffffu) << 16)); // dim0 hi | tensor_dim1 lo
    g1[3] = (int)(((rr >> 16) & 0xffffu) | (tile0 << 16));           // dim1 hi | tile_dim0
    g1[4] = (int)(rr & 0xffffu);                    // tile_dim1 = rows, tile_dim2=0
    g1[5] = (int)96;                                // tensor_dim0_stride[31:0] = 96 elems
    i32x4 z4 = {0, 0, 0, 0};
#if __clang_major__ >= 23
    i32x8 z8 = {0, 0, 0, 0, 0, 0, 0, 0};
    __builtin_amdgcn_tensor_load_to_lds(g0, g1, z4, z4, z8, 0);
#else
    __builtin_amdgcn_tensor_load_to_lds(g0, g1, z4, z4, 0);
#endif
}

// --------------------------- selective scan --------------------------------
// One thread per (b, d) channel; 16-wide state in VGPRs. B/C vectors for
// CHUNK steps are staged in LDS by the TDM (double-buffered, async), so the
// serial loop only touches LDS + coalesced dt/u/z reads. Fuses D-skip and
// SiLU(z) gate; emits gated y as f16 for the out-proj WMMA.
__global__ void scan_kernel(const float* __restrict__ xdbl,
                            const float* __restrict__ dt,
                            const float* __restrict__ uconv,
                            const float* __restrict__ z,
                            const float* __restrict__ A_log,
                            const float* __restrict__ D_param,
                            _Float16* __restrict__ y_h) {
    const int b = blockIdx.y;
    const int d = blockIdx.x * blockDim.x + threadIdx.x;   // 0..DINNER-1

    float Ad[DSTATE];
#pragma unroll
    for (int n = 0; n < DSTATE; ++n)
        Ad[n] = -__expf(A_log[(size_t)d * DSTATE + n]);
    const float Dp = D_param[d];

    float h[DSTATE];
#pragma unroll
    for (int n = 0; n < DSTATE; ++n) h[n] = 0.f;

    __shared__ float sBC[2][CHUNK][2 * DSTATE];
    const size_t tok0 = (size_t)b * NL;

    // prologue: async-stage chunk 0
    tdm_load_bc(xdbl + tok0 * DX + DTRANK,
                (unsigned)(size_t)&sBC[0][0][0], CHUNK);

    for (int c = 0; c < NL / CHUNK; ++c) {
        const int buf = c & 1;
        __builtin_amdgcn_s_wait_tensorcnt(0);  // our staged chunk has landed
        __syncthreads();                       // everyone's has landed
        if (c + 1 < NL / CHUNK)                // async-stage next chunk
            tdm_load_bc(xdbl + (tok0 + (size_t)(c + 1) * CHUNK) * DX + DTRANK,
                        (unsigned)(size_t)&sBC[buf ^ 1][0][0], CHUNK);

#pragma unroll
        for (int j = 0; j < CHUNK; ++j) {
            const size_t t = tok0 + (size_t)c * CHUNK + j;
            const float* bc = sBC[buf][j];
            const float dtv = dt[t * DINNER + d];
            const float uv  = uconv[t * DINNER + d];
            const float du  = dtv * uv;
            float acc = 0.f;
#pragma unroll
            for (int n = 0; n < DSTATE; ++n) {
                float dA = __expf(dtv * Ad[n]);
                h[n] = dA * h[n] + du * bc[n];
                acc += h[n] * bc[DSTATE + n];
            }
            const float yraw = acc + uv * Dp;
            const float zv   = z[t * DINNER + d];
            const float gate = zv / (1.f + __expf(-zv));
            y_h[t * DINNER + d] = (_Float16)(yraw * gate);
        }
        __syncthreads();                       // done reading buf before reuse
    }
}

// ------------------------------- launch ------------------------------------

extern "C" void kernel_launch(void* const* d_in, const int* in_sizes, int n_in,
                              void* d_out, int out_size, void* d_ws, size_t ws_size,
                              hipStream_t stream) {
    (void)in_sizes; (void)n_in; (void)out_size; (void)ws_size;

    const float* x       = (const float*)d_in[0];
    const float* gamma   = (const float*)d_in[1];
    const float* beta    = (const float*)d_in[2];
    const float* W_in    = (const float*)d_in[3];
    const float* conv_w  = (const float*)d_in[4];
    const float* conv_b  = (const float*)d_in[5];
    const float* W_x     = (const float*)d_in[6];
    const float* W_dt    = (const float*)d_in[7];
    const float* b_dt    = (const float*)d_in[8];
    const float* A_log   = (const float*)d_in[9];
    const float* D_param = (const float*)d_in[10];
    const float* W_out   = (const float*)d_in[11];
    float* out = (float*)d_out;

    // ---- workspace layout ----
    char* ws = (char*)d_ws;
    size_t off = 0;
    auto alloc = [&](size_t bytes) { char* p = ws + off; off += (bytes + 255) & ~(size_t)255; return p; };

    _Float16* Win_h  = (_Float16*)alloc((size_t)2 * DINNER * DMODEL * 2);
    _Float16* Wx_h   = (_Float16*)alloc((size_t)DX * DINNER * 2);
    _Float16* Wdt_h  = (_Float16*)alloc((size_t)DINNER * DTRANK * 2);
    _Float16* Wout_h = (_Float16*)alloc((size_t)DMODEL * DINNER * 2);
    _Float16* xn_h   = (_Float16*)alloc((size_t)NTOK * DMODEL * 2);
    float*    u_f    = (float*)   alloc((size_t)NTOK * DINNER * 4);
    float*    z_f    = (float*)   alloc((size_t)NTOK * DINNER * 4);
    float*    ucv_f  = (float*)   alloc((size_t)NTOK * DINNER * 4);
    _Float16* ucv_h  = (_Float16*)alloc((size_t)NTOK * DINNER * 2);
    float*    xdbl   = (float*)   alloc((size_t)NTOK * DX * 4);
    _Float16* dtr_h  = (_Float16*)alloc((size_t)NTOK * DTRANK * 2);
    float*    dt_f   = (float*)   alloc((size_t)NTOK * DINNER * 4);
    _Float16* y_h    = (_Float16*)alloc((size_t)NTOK * DINNER * 2);

    auto cdiv = [](long a, long b) { return (unsigned)((a + b - 1) / b); };

    // 1) weight down-casts (resident in 192MB L2 for the whole dispatch)
    cast_f32_to_f16_kernel<<<cdiv(2L*DINNER*DMODEL,256),256,0,stream>>>(W_in,  Win_h,  2*DINNER*DMODEL);
    cast_f32_to_f16_kernel<<<cdiv((long)DX*DINNER,256),  256,0,stream>>>(W_x,   Wx_h,   DX*DINNER);
    cast_f32_to_f16_kernel<<<cdiv((long)DINNER*DTRANK,256),256,0,stream>>>(W_dt, Wdt_h, DINNER*DTRANK);
    cast_f32_to_f16_kernel<<<cdiv((long)DMODEL*DINNER,256),256,0,stream>>>(W_out,Wout_h,DMODEL*DINNER);

    // 2) layernorm -> f16
    layernorm_kernel<<<NTOK, 256, 0, stream>>>(x, gamma, beta, xn_h);

    // 3) in-proj: u = xn @ W_in[0:DI]^T ; z = xn @ W_in[DI:2DI]^T
    {
        unsigned grp = (NTOK/32) * (DINNER/64);
        gemm_wmma_kernel<2,4,DINNER,DMODEL><<<cdiv(grp,8),256,0,stream>>>(
            xn_h, Win_h, u_f, NTOK, nullptr);
        gemm_wmma_kernel<2,4,DINNER,DMODEL><<<cdiv(grp,8),256,0,stream>>>(
            xn_h, Win_h + (size_t)DINNER*DMODEL, z_f, NTOK, nullptr);
    }

    // 4) causal depthwise conv + SiLU
    conv_silu_kernel<<<cdiv((long)NTOK*DINNER,256),256,0,stream>>>(u_f, conv_w, conv_b, ucv_f, ucv_h);

    // 5) x-proj: x_dbl = u_conv @ W_x^T  (N=96 -> 32x48 tiles)
    {
        unsigned grp = (NTOK/32) * (DX/48);
        gemm_wmma_kernel<2,3,DX,DINNER><<<cdiv(grp,8),256,0,stream>>>(
            ucv_h, Wx_h, xdbl, NTOK, nullptr);
    }

    // 6) dt-proj: dt = softplus(dt_r @ W_dt^T + b_dt)
    dtr_cast_kernel<<<cdiv((long)NTOK*DTRANK,256),256,0,stream>>>(xdbl, dtr_h);
    {
        unsigned grp = (NTOK/32) * (DINNER/64);
        gemm_wmma_kernel<2,4,DINNER,DTRANK><<<cdiv(grp,8),256,0,stream>>>(
            dtr_h, Wdt_h, dt_f, NTOK, nullptr);
    }
    softplus_kernel<<<cdiv((long)NTOK*DINNER,256),256,0,stream>>>(dt_f, b_dt);

    // 7) selective scan: TDM-staged B/C, fused D-skip + SiLU(z) gate, y -> f16
    scan_kernel<<<dim3(DINNER/256, NB), 256, 0, stream>>>(
        xdbl, dt_f, ucv_f, z_f, A_log, D_param, y_h);

    // 8) out-proj with fused residual: out = x + y @ W_out^T
    {
        unsigned grp = (NTOK/32) * (DMODEL/64);
        gemm_wmma_kernel<2,4,DMODEL,DINNER><<<cdiv(grp,8),256,0,stream>>>(
            y_h, Wout_h, out, NTOK, /*addend=*/x);
    }
}
